// ScaledDotProductAttention_9680856285905
// MI455X (gfx1250) — compile-verified, matrix operand order checked by reading, not run on hardware
//
#include <hip/hip_runtime.h>

#define BNH   32
#define LQ    1024
#define DK    64
#define RNUM  129
#define RPAD  132
#define CLIPV 64

typedef __attribute__((ext_vector_type(16))) __bf16 v16bf;
typedef __attribute__((ext_vector_type(8)))  float  v8f;
typedef unsigned int u32x4 __attribute__((ext_vector_type(4)));
typedef int          i32x8 __attribute__((ext_vector_type(8)));
typedef int          i32x4 __attribute__((ext_vector_type(4)));

#if __has_builtin(__builtin_amdgcn_tensor_load_to_lds) && __has_builtin(__builtin_amdgcn_tensor_store_from_lds)
#define HAVE_TDM 1
#else
#define HAVE_TDM 0
#endif

union BFU { unsigned short u; __bf16 b; };

__device__ __forceinline__ unsigned short f2bf(float f) {
    unsigned int u = __float_as_uint(f);
    unsigned int r = u + 0x7FFFu + ((u >> 16) & 1u);   // round-to-nearest-even
    return (unsigned short)(r >> 16);
}

// K-offset inside a 16x16x32 bf16 WMMA operand for vector element e, lane-half g.
__device__ __forceinline__ int koff(int e, int g) {
    return (e < 8) ? (g * 8 + e) : (16 + g * 8 + (e - 8));
}

// Low 32 bits of a flat address to an LDS object = workgroup-relative LDS byte offset.
__device__ __forceinline__ unsigned lds_off(const void* p) {
    return (unsigned)(unsigned long long)p;
}

#if HAVE_TDM
// 2D f32 tile descriptor: tile1 rows x tile0 cols, row stride stride0 (elements).
// D# group0: count=1 | lds_addr | global_addr[56:0] | type=2.
// D# group1: data_size=2 (4B); tensor_dim0/1 = 1<<20 (never OOB); tile dims; dim0 stride.
__device__ __forceinline__ void tdm_build(u32x4& g0, i32x8& g1,
                                          unsigned lds, const void* gptr,
                                          unsigned tile0, unsigned tile1, unsigned stride0) {
    unsigned long long ga = (unsigned long long)gptr;
    g0[0] = 1u;
    g0[1] = lds;
    g0[2] = (unsigned)ga;
    g0[3] = (unsigned)((ga >> 32) & 0x1FFFFFFull) | 0x80000000u;  // ga[56:32] | type=2
    g1[0] = 0x00020000;                       // wg_mask=0, data_size=4B
    g1[1] = 0;                                // tensor_dim0[15:0] (dim0 = 0x100000)
    g1[2] = 0x0010;                           // tensor_dim0[31:16] | tensor_dim1[15:0]
    g1[3] = (int)((tile0 << 16) | 0x0010u);   // tensor_dim1[31:16] | tile_dim0
    g1[4] = (int)tile1;                       // tile_dim1 | tile_dim2=0
    g1[5] = (int)stride0;                     // tensor_dim0_stride[31:0]
    g1[6] = 0;
    g1[7] = 0;
}
__device__ __forceinline__ void tdm_load_2d(unsigned lds, const void* gptr,
                                            unsigned tile0, unsigned tile1, unsigned stride0) {
    u32x4 g0; i32x8 g1;
    i32x4 gz  = {0, 0, 0, 0};
    i32x8 gz8 = {0, 0, 0, 0, 0, 0, 0, 0};
    tdm_build(g0, g1, lds, gptr, tile0, tile1, stride0);
    __builtin_amdgcn_tensor_load_to_lds(g0, g1, gz, gz, gz8, 0);
}
__device__ __forceinline__ void tdm_store_2d(unsigned lds, const void* gptr,
                                             unsigned tile0, unsigned tile1, unsigned stride0) {
    u32x4 g0; i32x8 g1;
    i32x4 gz  = {0, 0, 0, 0};
    i32x8 gz8 = {0, 0, 0, 0, 0, 0, 0, 0};
    tdm_build(g0, g1, lds, gptr, tile0, tile1, stride0);
    __builtin_amdgcn_tensor_store_from_lds(g0, g1, gz, gz, gz8, 0);
}
#endif

__global__ __launch_bounds__(256) void relpos_attn_kernel(
    const float* __restrict__ q,  const float* __restrict__ k,
    const float* __restrict__ v,  const float* __restrict__ attn_w,
    const float* __restrict__ emb_k, const float* __restrict__ emb_v,
    const int* __restrict__ pos,  const unsigned char* __restrict__ mask,
    float* __restrict__ out, float* __restrict__ p)
{
    __shared__ float S[16][LQ];                         // 64 KB: logits -> exp -> probs
    __shared__ __align__(16) unsigned char pool[32768]; // phase B: kbf(16K)+awbuf(16K); PV: vT(17K)
    __shared__ unsigned short qbf[16][DK];
    __shared__ float qs[16][DK];
    __shared__ float qe[16][RPAD];
    __shared__ float av[16][RPAD];
    __shared__ float outbuf[16][DK];
    __shared__ int   posj[LQ];
    __shared__ float red[16][16];
    __shared__ float rowmax[16], rowrcp[16];
    __shared__ int   posi[16];

    unsigned short (*kbf)[DK]  = (unsigned short (*)[DK])pool;          // [128][64] bf16 bits
    float (*awbuf)[16][128]    = (float (*)[16][128])(pool + 16384);    // [2][16][128] f32
    unsigned short (*vT)[136]  = (unsigned short (*)[136])pool;         // [64][136] bf16 bits (PV)

    const int b   = blockIdx.y;
    const int i0  = blockIdx.x * 16;
    const int tid = threadIdx.x;
    const int w    = tid >> 5;
    const int lane = tid & 31;
    const int g    = lane >> 4;
    const int nl   = lane & 15;

    // ---- stage q tile, positions; zero accumulators ----
    for (int e = tid; e < 16 * DK; e += 256) {
        int m = e >> 6, d = e & 63;
        float qv = q[((size_t)b * LQ + (i0 + m)) * DK + d];
        qs[m][d]  = qv;
        qbf[m][d] = f2bf(qv);
    }
    for (int e = tid; e < LQ; e += 256) posj[e] = pos[b * LQ + e];
    if (tid < 16) posi[tid] = pos[b * LQ + i0 + tid];
    for (int e = tid; e < 16 * RPAD; e += 256) (&av[0][0])[e] = 0.f;
    for (int e = tid; e < 16 * DK;   e += 256) (&outbuf[0][0])[e] = 0.f;

#if HAVE_TDM
    // prefetch attn_w tile jb=0 via the Tensor Data Mover (one issue per workgroup)
    if (w == 0)
        tdm_load_2d(lds_off(&awbuf[0][0][0]),
                    attn_w + ((size_t)b * LQ + i0) * LQ, 128u, 16u, (unsigned)LQ);
#endif
    __syncthreads();

    // ---- qe[i][r] = q[i] . emb_k[r]   (16 x 129, scalar f32) ----
    for (int e = tid; e < 16 * RNUM; e += 256) {
        int m = e / RNUM, r = e % RNUM;
        const float* er = emb_k + r * DK;
        float acc = 0.f;
        #pragma unroll 8
        for (int d = 0; d < DK; ++d) acc += qs[m][d] * er[d];
        qe[m][r] = acc;
    }
    __syncthreads();

    // ---- logits: S = (Q K^T + gather(qe) + attn_w) / 8, masked ----
    for (int jb = 0; jb < 8; ++jb) {
#if HAVE_TDM
        if (w == 0) {                // prefetch next attn_w tile, wait for current
            if (jb + 1 < 8) {
                tdm_load_2d(lds_off(&awbuf[(jb + 1) & 1][0][0]),
                            attn_w + ((size_t)b * LQ + i0) * LQ + (jb + 1) * 128,
                            128u, 16u, (unsigned)LQ);
                __builtin_amdgcn_s_wait_tensorcnt(1);
            } else {
                __builtin_amdgcn_s_wait_tensorcnt(0);
            }
        }
#else
        for (int e = tid; e < 16 * 128; e += 256) {
            int m = e >> 7, jl = e & 127;
            awbuf[0][m][jl] = attn_w[((size_t)b * LQ + (i0 + m)) * LQ + jb * 128 + jl];
        }
#endif
        for (int e = tid; e < 128 * DK; e += 256) {
            int jl = e >> 6, d = e & 63;
            kbf[jl][d] = f2bf(k[((size_t)b * LQ + jb * 128 + jl) * DK + d]);
        }
        __syncthreads();
        {
            const int j0 = jb * 128 + w * 16;
            v8f acc = {};
            #pragma unroll
            for (int s = 0; s < 2; ++s) {
                v16bf a, bb;
                #pragma unroll
                for (int e = 0; e < 16; ++e) {
                    int kk = s * 32 + koff(e, g);
                    BFU ca; ca.u = qbf[nl][kk];          a[e]  = ca.b;
                    BFU cb; cb.u = kbf[w * 16 + nl][kk]; bb[e] = cb.b;
                }
                acc = __builtin_amdgcn_wmma_f32_16x16x32_bf16(
                          false, a, false, bb, (short)0, acc, false, false);
            }
            #pragma unroll
            for (int idx = 0; idx < 8; ++idx) {
                int m = idx + 8 * g;
                int j = j0 + nl;
                int dd = posj[j] - posi[m];
                dd = dd < -CLIPV ? -CLIPV : (dd > CLIPV ? CLIPV : dd);
#if HAVE_TDM
                float aw = awbuf[jb & 1][m][w * 16 + nl];
#else
                float aw = awbuf[0][m][w * 16 + nl];
#endif
                float val = (acc[idx] + qe[m][dd + CLIPV] + aw) * 0.125f;
                if (mask[((size_t)b * LQ + (i0 + m)) * LQ + j]) val = -__builtin_inff();
                S[m][j] = val;
            }
        }
        __syncthreads();
    }

    // ---- row softmax over 1024 (16-thread teams per row) ----
    {
        int m = tid >> 4, tt = tid & 15;
        float mx = -__builtin_inff();
        for (int j = tt; j < LQ; j += 16) mx = fmaxf(mx, S[m][j]);
        red[m][tt] = mx;
        __syncthreads();
        if (tt == 0) {
            float r0 = red[m][0];
            #pragma unroll
            for (int x = 1; x < 16; ++x) r0 = fmaxf(r0, red[m][x]);
            rowmax[m] = r0;
        }
        __syncthreads();
        float rm  = rowmax[m];
        float sum = 0.f;
        for (int j = tt; j < LQ; j += 16) {
            float ex = __expf(S[m][j] - rm);
            S[m][j] = ex;
            sum += ex;
        }
        red[m][tt] = sum;
        __syncthreads();
        if (tt == 0) {
            float r0 = 0.f;
            #pragma unroll
            for (int x = 0; x < 16; ++x) r0 += red[m][x];
            rowrcp[m] = 1.0f / r0;
        }
        __syncthreads();
    }

    // ---- normalize in LDS, scatter probs into R buckets ----
    for (int e = tid; e < 16 * LQ; e += 256) {
        int m = e >> 10, j = e & 1023;
        float pv = S[m][j] * rowrcp[m];
        S[m][j] = pv;
        int dd = posj[j] - posi[m];
        dd = dd < -CLIPV ? -CLIPV : (dd > CLIPV ? CLIPV : dd);
        atomicAdd(&av[m][dd + CLIPV], pv);
    }
    __syncthreads();

#if HAVE_TDM
    // DMA the finished 16x1024 probability tile straight from LDS to p (overlaps PV)
    if (w == 0)
        tdm_store_2d(lds_off(&S[0][0]),
                     p + ((size_t)b * LQ + i0) * LQ, (unsigned)LQ, 16u, (unsigned)LQ);
#else
    for (int e = tid; e < 16 * LQ; e += 256) {
        int m = e >> 10, j = e & 1023;
        p[((size_t)b * LQ + (i0 + m)) * LQ + j] = S[m][j];
    }
#endif

    // ---- out = P @ V : wave owns (d-tile = w&3, K-half = w>>2) of each 128-j block ----
    {
        const int dt = w & 3, h = w >> 2;
        v8f acc = {};
        for (int jb = 0; jb < 8; ++jb) {
            // cooperative transposed bf16 stage of v block (coalesced global reads)
            for (int e = tid; e < 128 * DK; e += 256) {
                int jl = e >> 6, d = e & 63;
                vT[d][jl] = f2bf(v[((size_t)b * LQ + jb * 128 + jl) * DK + d]);
            }
            __syncthreads();
            #pragma unroll
            for (int s = 0; s < 2; ++s) {
                const int kl = h * 64 + s * 32;
                v16bf a, bb;
                #pragma unroll
                for (int e = 0; e < 16; ++e) {
                    int kk = kl + koff(e, g);
                    BFU ca; ca.u = f2bf(S[nl][jb * 128 + kk]); a[e]  = ca.b;
                    BFU cb; cb.u = vT[dt * 16 + nl][kk];       bb[e] = cb.b;
                }
                acc = __builtin_amdgcn_wmma_f32_16x16x32_bf16(
                          false, a, false, bb, (short)0, acc, false, false);
            }
            __syncthreads();
        }
        #pragma unroll
        for (int idx = 0; idx < 8; ++idx)
            atomicAdd(&outbuf[idx + 8 * g][dt * 16 + nl], acc[idx]);
    }
    __syncthreads();

    // ---- out += av @ emb_v ; store ----
    for (int e = tid; e < 16 * DK; e += 256) {
        int m = e >> 6, d = e & 63;
        float acc = outbuf[m][d];
        for (int r = 0; r < RNUM; ++r) acc += av[m][r] * emb_v[r * DK + d];
        out[((size_t)b * LQ + (i0 + m)) * DK + d] = acc;
    }

#if HAVE_TDM
    if (w == 0) __builtin_amdgcn_s_wait_tensorcnt(0);  // p DMA must land before wave exit
#endif
}

extern "C" void kernel_launch(void* const* d_in, const int* in_sizes, int n_in,
                              void* d_out, int out_size, void* d_ws, size_t ws_size,
                              hipStream_t stream) {
    (void)in_sizes; (void)n_in; (void)out_size; (void)d_ws; (void)ws_size;
    const float* q      = (const float*)d_in[0];
    const float* k      = (const float*)d_in[1];
    const float* v      = (const float*)d_in[2];
    const float* attn_w = (const float*)d_in[3];
    const float* emb_k  = (const float*)d_in[4];
    const float* emb_v  = (const float*)d_in[5];
    const int*   pos    = (const int*)d_in[6];
    const unsigned char* mask = (const unsigned char*)d_in[7];

    float* out = (float*)d_out;                       // (32,1024,64)
    float* p   = out + (size_t)BNH * LQ * DK;         // (32,1024,1024)

    dim3 grid(LQ / 16, BNH);
    relpos_attn_kernel<<<grid, 256, 0, stream>>>(q, k, v, attn_w, emb_k, emb_v,
                                                 pos, mask, out, p);
}
